// MultiHeadSelfAttentionModule_87643102642313
// MI455X (gfx1250) — matrix operand order repaired and checked by reading, not compiled, not run on hardware
//
#include <hip/hip_runtime.h>

#define BATCH  4
#define SEQ    1024
#define DMODEL 1024
#define NHEAD  16
#define DHEAD  64

typedef __attribute__((ext_vector_type(16))) _Float16 v16h;
typedef __attribute__((ext_vector_type(8)))  float    v8f;

#if __has_builtin(__builtin_amdgcn_s_wait_asynccnt)
#define WAIT_ASYNC(n) __builtin_amdgcn_s_wait_asynccnt(n)
#else
#define WAIT_ASYNC(n) asm volatile("s_wait_asynccnt %0" ::"i"(n) : "memory")
#endif

// ---------------- WMMA fragment helpers (v_wmma_f32_16x16x32_f16) ----------------
// A (16x32 f16, row-major source): lanes 0-15 -> rows 0-15, K {0..7,16..23};
//                                  lanes 16-31 -> rows 0-15, K {8..15,24..31}
__device__ __forceinline__ v16h load_a_rm(const _Float16* __restrict__ base, int ld, int lane) {
  const _Float16* p = base + (size_t)(lane & 15) * ld + ((lane < 16) ? 0 : 8);
  v16h a;
#pragma unroll
  for (int i = 0; i < 8; ++i) a[i] = p[i];
#pragma unroll
  for (int i = 0; i < 8; ++i) a[i + 8] = p[16 + i];
  return a;
}

// B (32x16 f16) from K-contiguous (transposed) storage: baseT points at BT[n0][k0].
__device__ __forceinline__ v16h load_b_tr(const _Float16* __restrict__ baseT, int ldT, int lane) {
  const _Float16* p = baseT + (size_t)(lane & 15) * ldT + ((lane < 16) ? 0 : 8);
  v16h b;
#pragma unroll
  for (int i = 0; i < 8; ++i) b[i] = p[i];
#pragma unroll
  for (int i = 0; i < 8; ++i) b[i + 8] = p[16 + i];
  return b;
}

// B fragment from an LDS-staged tile laid out [64 rows (n)][32 halfs (k)]
__device__ __forceinline__ v16h load_b_lds(const _Float16* buf, int t, int lane) {
  const _Float16* p = buf + (size_t)(t * 16 + (lane & 15)) * 32 + ((lane < 16) ? 0 : 8);
  v16h b;
#pragma unroll
  for (int i = 0; i < 8; ++i) b[i] = p[i];
#pragma unroll
  for (int i = 0; i < 8; ++i) b[i + 8] = p[16 + i];
  return b;
}

__device__ __forceinline__ v8f wmma_f16(v16h a, v16h b, v8f c) {
  return __builtin_amdgcn_wmma_f32_16x16x32_f16(false, a, false, b, (short)0, c, false, false);
}

// C/D (16x16 f32): lanes 0-15 -> N=lane, rows 0-7; lanes 16-31 -> rows 8-15.
__device__ __forceinline__ void store_c_f32(float* __restrict__ base, int ldc, int lane, v8f c) {
  float* p = base + (size_t)((lane < 16) ? 0 : 8) * ldc + (lane & 15);
#pragma unroll
  for (int r = 0; r < 8; ++r) p[(size_t)r * ldc] = c[r];
}
__device__ __forceinline__ void store_c_f16(_Float16* __restrict__ base, int ldc, int lane, v8f c) {
  _Float16* p = base + (size_t)((lane < 16) ? 0 : 8) * ldc + (lane & 15);
#pragma unroll
  for (int r = 0; r < 8; ++r) p[(size_t)r * ldc] = (_Float16)c[r];
}

// ---------------- async global -> LDS staging of one 64x32-f16 B tile ----------------
// Source rows have stride 1024 halfs (true for all staged GEMMs here).
// Wave `wave` copies rows [wave*8, wave*8+8): 4 lanes x 16B per row.
__device__ __forceinline__ void async_fill_b(const _Float16* BT_base, uint32_t lds_off,
                                             int wave, int lane) {
  const char* g = (const char*)BT_base + (size_t)wave * 8 * (1024 * 2);
  uint32_t voff = (uint32_t)((lane >> 2) * (1024 * 2) + (lane & 3) * 16);
  uint32_t loff = lds_off + (uint32_t)(wave * 8 * 64 + (lane >> 2) * 64 + (lane & 3) * 16);
  asm volatile("global_load_async_to_lds_b128 %0, %1, %2"
               :: "v"(loff), "v"(voff), "s"((uint64_t)g)
               : "memory");
}

// ---------------- elementwise / prep kernels ----------------
__global__ void wtrans_kernel(const float* __restrict__ W, _Float16* __restrict__ WT) {
  int tid = blockIdx.x * 256 + threadIdx.x;            // over DMODEL*DMODEL
  int k = tid >> 10, n = tid & 1023;
  WT[(size_t)n * DMODEL + k] = (_Float16)W[tid];
}

__global__ void f2h_kernel(const float* __restrict__ src, _Float16* __restrict__ dst) {
  size_t tid = (size_t)blockIdx.x * 256 + threadIdx.x;
  dst[tid] = (_Float16)src[tid];
}

__global__ void ln_kernel(const float* __restrict__ x, const float* __restrict__ g,
                          const float* __restrict__ b, _Float16* __restrict__ y) {
  const int row = blockIdx.x;                          // BATCH*SEQ rows
  const float* xr = x + (size_t)row * DMODEL;
  __shared__ float red[256];
  float v[4];
  float s = 0.f;
#pragma unroll
  for (int i = 0; i < 4; ++i) { v[i] = xr[threadIdx.x + i * 256]; s += v[i]; }
  red[threadIdx.x] = s; __syncthreads();
  for (int off = 128; off > 0; off >>= 1) {
    if (threadIdx.x < off) red[threadIdx.x] += red[threadIdx.x + off];
    __syncthreads();
  }
  const float mu = red[0] * (1.f / DMODEL);
  __syncthreads();
  float vs = 0.f;
#pragma unroll
  for (int i = 0; i < 4; ++i) { float d = v[i] - mu; vs += d * d; }
  red[threadIdx.x] = vs; __syncthreads();
  for (int off = 128; off > 0; off >>= 1) {
    if (threadIdx.x < off) red[threadIdx.x] += red[threadIdx.x + off];
    __syncthreads();
  }
  const float rstd = rsqrtf(red[0] * (1.f / DMODEL) + 1e-5f);
  _Float16* yr = y + (size_t)row * DMODEL;
#pragma unroll
  for (int i = 0; i < 4; ++i) {
    int c = threadIdx.x + i * 256;
    yr[c] = (_Float16)((v[i] - mu) * rstd * g[c] + b[c]);
  }
}

// ---------------- projection GEMM: C[4096,1024] = A*B (+bias), f16 out ----------------
// B tiles double-buffered in LDS via async copies. vmode==1: store V as [B,H,DH,S].
__global__ void gemm_proj_kernel(const _Float16* __restrict__ A, const _Float16* __restrict__ BT,
                                 const float* __restrict__ bias, _Float16* __restrict__ C,
                                 int vmode) {
  __shared__ _Float16 ldsB[2][64 * 32];
  const int lane = threadIdx.x & 31;
  const int wave = threadIdx.x >> 5;
  const int m0 = blockIdx.x * 128 + wave * 16;
  const int n0 = blockIdx.y * 64;
  const uint32_t lb[2] = { (uint32_t)(size_t)&ldsB[0][0], (uint32_t)(size_t)&ldsB[1][0] };
  v8f acc[4] = {};
  const _Float16* Ap = A + (size_t)m0 * DMODEL;
  const _Float16* Bp = BT + (size_t)n0 * DMODEL;

  async_fill_b(Bp, lb[0], wave, lane);
  for (int i = 0; i < 32; ++i) {
    const int k0 = i * 32;
    if (i + 1 < 32) {
      async_fill_b(Bp + (k0 + 32), lb[(i + 1) & 1], wave, lane);
      WAIT_ASYNC(1);
    } else {
      WAIT_ASYNC(0);
    }
    __syncthreads();
    if (k0 + 32 < DMODEL) __builtin_prefetch(Ap + k0 + 32, 0, 3);
    v16h a = load_a_rm(Ap + k0, DMODEL, lane);
    const _Float16* bbuf = &ldsB[i & 1][0];
#pragma unroll
    for (int t = 0; t < 4; ++t) {
      v16h b = load_b_lds(bbuf, t, lane);
      acc[t] = wmma_f16(a, b, acc[t]);
    }
    __syncthreads();
  }

  const int rb = (lane < 16) ? 0 : 8;
  if (vmode == 0) {
#pragma unroll
    for (int t = 0; t < 4; ++t) {
      int n = n0 + t * 16 + (lane & 15);
      float bv = bias ? bias[n] : 0.f;
      _Float16* p = C + (size_t)(m0 + rb) * DMODEL + n;
#pragma unroll
      for (int r = 0; r < 8; ++r) p[(size_t)r * DMODEL] = (_Float16)(acc[t][r] + bv);
    }
  } else {
#pragma unroll
    for (int t = 0; t < 4; ++t) {
      int n = n0 + t * 16 + (lane & 15);
      int h = n >> 6, dd = n & 63;
      float bv = bias ? bias[n] : 0.f;
#pragma unroll
      for (int r = 0; r < 8; ++r) {
        int m = m0 + rb + r;
        int bb = m >> 10, s = m & 1023;
        C[(((size_t)bb * NHEAD + h) * DHEAD + dd) * SEQ + s] = (_Float16)(acc[t][r] + bv);
      }
    }
  }
}

// ---------------- content + pos scores (per batch): cs/ps [H,S,S] f32 ----------------
__global__ void score_kernel(const _Float16* __restrict__ Q, const _Float16* __restrict__ K,
                             const _Float16* __restrict__ P,
                             const float* __restrict__ ubias, const float* __restrict__ vbias,
                             float* __restrict__ cs, float* __restrict__ ps) {
  const int lane = threadIdx.x & 31;
  const int wave = threadIdx.x >> 5;
  const int q0 = blockIdx.x * 128 + wave * 16;
  const int n0 = blockIdx.y * 64;                      // key block
  const int h  = blockIdx.z;
  v8f accC[4] = {};
  v8f accP[4] = {};
  const int arow = q0 + (lane & 15);
  const int kb = (lane < 16) ? 0 : 8;
#pragma unroll
  for (int d0 = 0; d0 < DHEAD; d0 += 32) {
    const _Float16* qp = Q + (size_t)arow * DMODEL + h * DHEAD + d0 + kb;
    const float* up = ubias + h * DHEAD + d0 + kb;
    const float* vp = vbias + h * DHEAD + d0 + kb;
    v16h aU, aV;
#pragma unroll
    for (int i = 0; i < 8; ++i) {
      float qv = (float)qp[i];
      aU[i] = (_Float16)(qv + up[i]);
      aV[i] = (_Float16)(qv + vp[i]);
    }
#pragma unroll
    for (int i = 0; i < 8; ++i) {
      float qv = (float)qp[16 + i];
      aU[8 + i] = (_Float16)(qv + up[16 + i]);
      aV[8 + i] = (_Float16)(qv + vp[16 + i]);
    }
#pragma unroll
    for (int t = 0; t < 4; ++t) {
      v16h bK = load_b_tr(K + (size_t)(n0 + t * 16) * DMODEL + h * DHEAD + d0, DMODEL, lane);
      accC[t] = wmma_f16(aU, bK, accC[t]);
      v16h bP = load_b_tr(P + (size_t)(n0 + t * 16) * DMODEL + h * DHEAD + d0, DMODEL, lane);
      accP[t] = wmma_f16(aV, bP, accP[t]);
    }
  }
#pragma unroll
  for (int t = 0; t < 4; ++t) {
    store_c_f32(cs + ((size_t)h * SEQ + q0) * SEQ + n0 + t * 16, SEQ, lane, accC[t]);
    store_c_f32(ps + ((size_t)h * SEQ + q0) * SEQ + n0 + t * 16, SEQ, lane, accP[t]);
  }
}

// ---------------- rel_shift gather + combine + 1/sqrt(D) scale (in place into cs) ----------------
__global__ void combine_kernel(const float* __restrict__ ps, float* __restrict__ cs) {
  size_t idx = (size_t)blockIdx.x * 256 + threadIdx.x;   // over NHEAD*SEQ*SEQ
  int j = (int)(idx & (SEQ - 1));
  int i = (int)((idx >> 10) & (SEQ - 1));
  int h = (int)(idx >> 20);
  size_t flat = (size_t)(i + 1) * SEQ + j;               // reference _rel_shift index math
  int a = (int)(flat / (SEQ + 1));
  int c = (int)(flat % (SEQ + 1));
  float shifted = (c == 0) ? 0.f : ps[((size_t)h * SEQ + a) * SEQ + (c - 1)];
  cs[idx] = (cs[idx] + shifted) * 0.03125f;              // 1/sqrt(1024)
}

// ---------------- row softmax -> f16 attn ----------------
__global__ void softmax_kernel(const float* __restrict__ sc, _Float16* __restrict__ attn) {
  const int row = blockIdx.x;                            // NHEAD*SEQ rows
  const float* sr = sc + (size_t)row * SEQ;
  __shared__ float red[256];
  float v[4];
  float m = -3.0e38f;
#pragma unroll
  for (int i = 0; i < 4; ++i) { v[i] = sr[threadIdx.x + i * 256]; m = fmaxf(m, v[i]); }
  red[threadIdx.x] = m; __syncthreads();
  for (int off = 128; off > 0; off >>= 1) {
    if (threadIdx.x < off) red[threadIdx.x] = fmaxf(red[threadIdx.x], red[threadIdx.x + off]);
    __syncthreads();
  }
  m = red[0];
  __syncthreads();
  float s = 0.f;
#pragma unroll
  for (int i = 0; i < 4; ++i) { v[i] = __expf(v[i] - m); s += v[i]; }
  red[threadIdx.x] = s; __syncthreads();
  for (int off = 128; off > 0; off >>= 1) {
    if (threadIdx.x < off) red[threadIdx.x] += red[threadIdx.x + off];
    __syncthreads();
  }
  const float inv = 1.0f / red[0];
  _Float16* ar = attn + (size_t)row * SEQ;
#pragma unroll
  for (int i = 0; i < 4; ++i) ar[threadIdx.x + i * 256] = (_Float16)(v[i] * inv);
}

// ---------------- ctx = attn @ V (per batch), f16 out [S, D] ----------------
__global__ void ctx_kernel(const _Float16* __restrict__ attn, const _Float16* __restrict__ Vt,
                           _Float16* __restrict__ ctx) {
  __shared__ _Float16 ldsB[2][64 * 32];
  const int lane = threadIdx.x & 31;
  const int wave = threadIdx.x >> 5;
  const int q0 = blockIdx.x * 128 + wave * 16;
  const int h  = blockIdx.y;
  const uint32_t lb[2] = { (uint32_t)(size_t)&ldsB[0][0], (uint32_t)(size_t)&ldsB[1][0] };
  v8f acc[4] = {};
  const _Float16* Ar = attn + ((size_t)h * SEQ + q0) * SEQ;
  const _Float16* Vh = Vt + (size_t)h * DHEAD * SEQ;     // [DHEAD=64 rows, SEQ] per head

  async_fill_b(Vh, lb[0], wave, lane);
  for (int i = 0; i < 32; ++i) {
    const int k0 = i * 32;
    if (i + 1 < 32) {
      async_fill_b(Vh + (k0 + 32), lb[(i + 1) & 1], wave, lane);
      WAIT_ASYNC(1);
    } else {
      WAIT_ASYNC(0);
    }
    __syncthreads();
    if (k0 + 32 < SEQ) __builtin_prefetch(Ar + k0 + 32, 0, 3);
    v16h a = load_a_rm(Ar + k0, SEQ, lane);
    const _Float16* bbuf = &ldsB[i & 1][0];
#pragma unroll
    for (int t = 0; t < 4; ++t) {
      v16h b = load_b_lds(bbuf, t, lane);
      acc[t] = wmma_f16(a, b, acc[t]);
    }
    __syncthreads();
  }
#pragma unroll
  for (int t = 0; t < 4; ++t)
    store_c_f16(ctx + (size_t)q0 * DMODEL + h * DHEAD + t * 16, DMODEL, lane, acc[t]);
}

// ---------------- output GEMM + bias + residual, f32 out ----------------
__global__ void gemm_out_kernel(const _Float16* __restrict__ A, const _Float16* __restrict__ BT,
                                const float* __restrict__ bias, const float* __restrict__ xres,
                                float* __restrict__ out) {
  __shared__ _Float16 ldsB[2][64 * 32];
  const int lane = threadIdx.x & 31;
  const int wave = threadIdx.x >> 5;
  const int m0 = blockIdx.x * 128 + wave * 16;
  const int n0 = blockIdx.y * 64;
  const uint32_t lb[2] = { (uint32_t)(size_t)&ldsB[0][0], (uint32_t)(size_t)&ldsB[1][0] };
  v8f acc[4] = {};
  const _Float16* Ap = A + (size_t)m0 * DMODEL;
  const _Float16* Bp = BT + (size_t)n0 * DMODEL;

  async_fill_b(Bp, lb[0], wave, lane);
  for (int i = 0; i < 32; ++i) {
    const int k0 = i * 32;
    if (i + 1 < 32) {
      async_fill_b(Bp + (k0 + 32), lb[(i + 1) & 1], wave, lane);
      WAIT_ASYNC(1);
    } else {
      WAIT_ASYNC(0);
    }
    __syncthreads();
    if (k0 + 32 < DMODEL) __builtin_prefetch(Ap + k0 + 32, 0, 3);
    v16h a = load_a_rm(Ap + k0, DMODEL, lane);
    const _Float16* bbuf = &ldsB[i & 1][0];
#pragma unroll
    for (int t = 0; t < 4; ++t) {
      v16h b = load_b_lds(bbuf, t, lane);
      acc[t] = wmma_f16(a, b, acc[t]);
    }
    __syncthreads();
  }

  const int rb = (lane < 16) ? 0 : 8;
#pragma unroll
  for (int t = 0; t < 4; ++t) {
    int n = n0 + t * 16 + (lane & 15);
    float bv = bias[n];
#pragma unroll
    for (int r = 0; r < 8; ++r) {
      size_t off = (size_t)(m0 + rb + r) * DMODEL + n;
      out[off] = acc[t][r] + bv + xres[off];
    }
  }
}

// ---------------- host ----------------
extern "C" void kernel_launch(void* const* d_in, const int* in_sizes, int n_in,
                              void* d_out, int out_size, void* d_ws, size_t ws_size,
                              hipStream_t stream) {
  (void)in_sizes; (void)n_in; (void)out_size; (void)ws_size;
  const float* x    = (const float*)d_in[0];
  const float* pos  = (const float*)d_in[1];
  const float* lng  = (const float*)d_in[2];
  const float* lnb  = (const float*)d_in[3];
  const float* Wq   = (const float*)d_in[4];
  const float* bq   = (const float*)d_in[5];
  const float* Wk   = (const float*)d_in[6];
  const float* bk   = (const float*)d_in[7];
  const float* Wv   = (const float*)d_in[8];
  const float* bv   = (const float*)d_in[9];
  const float* Wpos = (const float*)d_in[10];
  const float* ub   = (const float*)d_in[11];
  const float* vb   = (const float*)d_in[12];
  const float* Wout = (const float*)d_in[13];
  const float* bout = (const float*)d_in[14];
  float* out = (float*)d_out;

  char* cur = (char*)d_ws;
  auto carve = [&](size_t bytes) -> void* {
    void* p = (void*)cur;
    cur += (bytes + 255) & ~(size_t)255;
    return p;
  };
  const size_t ND  = (size_t)BATCH * SEQ * DMODEL;     // 4Mi elements
  const size_t WSZ = (size_t)DMODEL * DMODEL;          // 1Mi elements
  const size_t SC  = (size_t)NHEAD * SEQ * SEQ;        // 16Mi elements (per batch)

  _Float16* y16   = (_Float16*)carve(ND * 2);
  _Float16* p16   = (_Float16*)carve(ND * 2);
  _Float16* wqT   = (_Float16*)carve(WSZ * 2);
  _Float16* wkT   = (_Float16*)carve(WSZ * 2);
  _Float16* wvT   = (_Float16*)carve(WSZ * 2);
  _Float16* wpT   = (_Float16*)carve(WSZ * 2);
  _Float16* woT   = (_Float16*)carve(WSZ * 2);
  _Float16* Qh    = (_Float16*)carve(ND * 2);
  _Float16* Kh    = (_Float16*)carve(ND * 2);
  _Float16* Ph    = (_Float16*)carve(ND * 2);
  _Float16* Vt    = (_Float16*)carve(ND * 2);          // [B, H, DH, S]
  _Float16* ctx16 = (_Float16*)carve(ND * 2);
  float*    cs    = (float*)carve(SC * 4);             // per-batch, L2-resident
  float*    ps    = (float*)carve(SC * 4);
  _Float16* attn  = (_Float16*)carve(SC * 2);

  wtrans_kernel<<<(unsigned)(WSZ / 256), 256, 0, stream>>>(Wq, wqT);
  wtrans_kernel<<<(unsigned)(WSZ / 256), 256, 0, stream>>>(Wk, wkT);
  wtrans_kernel<<<(unsigned)(WSZ / 256), 256, 0, stream>>>(Wv, wvT);
  wtrans_kernel<<<(unsigned)(WSZ / 256), 256, 0, stream>>>(Wpos, wpT);
  wtrans_kernel<<<(unsigned)(WSZ / 256), 256, 0, stream>>>(Wout, woT);
  f2h_kernel<<<(unsigned)(ND / 256), 256, 0, stream>>>(pos, p16);
  ln_kernel<<<BATCH * SEQ, 256, 0, stream>>>(x, lng, lnb, y16);

  dim3 gp(32, 16);   // M/128 x N/64
  gemm_proj_kernel<<<gp, 256, 0, stream>>>(y16, wqT, bq, Qh, 0);
  gemm_proj_kernel<<<gp, 256, 0, stream>>>(y16, wkT, bk, Kh, 0);
  gemm_proj_kernel<<<gp, 256, 0, stream>>>(y16, wvT, bv, Vt, 1);
  gemm_proj_kernel<<<gp, 256, 0, stream>>>(p16, wpT, nullptr, Ph, 0);

  for (int b = 0; b < BATCH; ++b) {
    const size_t bo = (size_t)b * SEQ * DMODEL;
    score_kernel<<<dim3(8, 16, 16), 256, 0, stream>>>(Qh + bo, Kh + bo, Ph + bo, ub, vb, cs, ps);
    combine_kernel<<<(unsigned)(SC / 256), 256, 0, stream>>>(ps, cs);
    softmax_kernel<<<NHEAD * SEQ, 256, 0, stream>>>(cs, attn);
    ctx_kernel<<<dim3(8, 16), 256, 0, stream>>>(attn, Vt + bo, ctx16 + bo);
  }

  gemm_out_kernel<<<gp, 256, 0, stream>>>(ctx16, woT, bout, x, out);
}
